// FusedLinearCrossEntropyLoss_27848567948093
// MI455X (gfx1250) — matrix operand order repaired and checked by reading, not compiled, not run on hardware
//
#include <hip/hip_runtime.h>
#include <hip/hip_bf16.h>
#include <math.h>

// ---------------- problem constants (from reference) ----------------
#define H       2048
#define V       131072
#define NROWS   4095          // B*S - 1
#define NPAD    4096
#define NCHUNK  32
#define CHUNKV  4096          // V / NCHUNK
#define MBLK    64            // rows per block (4 WMMA row tiles)
#define ROWGRPS (NPAD / MBLK) // 64
#define COLTILES_PER_CHUNK (CHUNKV / 16) // 256
#define IGNORE_INDEX (-100)

typedef __attribute__((ext_vector_type(16))) __bf16 v16bf;
typedef __attribute__((ext_vector_type(8)))  float  v8f;

__device__ __forceinline__ unsigned short f32_to_bf16_rne(float f) {
  unsigned u = __float_as_uint(f);
  u += 0x7FFFu + ((u >> 16) & 1u);   // round-to-nearest-even
  return (unsigned short)(u >> 16);
}

// online softmax: fold one new logit v into running (m, s)
__device__ __forceinline__ void online_merge(float v, float& m, float& s) {
  const float mn = fmaxf(m, v);
  s = s * __expf(m - mn) + __expf(v - mn);
  m = mn;
}
// merge two (m, s) pairs
__device__ __forceinline__ void pair_merge(float& m, float& s, float om, float os) {
  const float mn = fmaxf(m, om);
  s = s * __expf(m - mn) + os * __expf(om - mn);
  m = mn;
}

// -------- kernel 0: zero the loss accumulators --------
__global__ void flce_init(float* scal) {
  if (threadIdx.x < 2) scal[threadIdx.x] = 0.0f;
}

// -------- kernel 1: weight fp32 -> bf16 (done once, streamed) --------
__global__ __launch_bounds__(256)
void flce_cvt_weight(const float* __restrict__ w, unsigned short* __restrict__ wbf) {
  const size_t total  = (size_t)V * H / 4;
  const size_t stride = (size_t)gridDim.x * blockDim.x;
  for (size_t g = (size_t)blockIdx.x * blockDim.x + threadIdx.x; g < total; g += stride) {
    float4 f = reinterpret_cast<const float4*>(w)[g];
    ushort4 o;
    o.x = f32_to_bf16_rne(f.x); o.y = f32_to_bf16_rne(f.y);
    o.z = f32_to_bf16_rne(f.z); o.w = f32_to_bf16_rne(f.w);
    reinterpret_cast<ushort4*>(wbf)[g] = o;
  }
}

// -------- kernel 2: hidden fp32 -> bf16, pad row 4095 with zeros --------
__global__ __launch_bounds__(256)
void flce_cvt_hidden(const float* __restrict__ h, unsigned short* __restrict__ hbf) {
  const size_t g = (size_t)blockIdx.x * blockDim.x + threadIdx.x;
  const size_t base = g * 4;
  if (base >= (size_t)NPAD * H) return;
  ushort4 o;
  if (base < (size_t)NROWS * H) {      // NROWS*H is divisible by 4
    float4 f = reinterpret_cast<const float4*>(h)[g];
    o.x = f32_to_bf16_rne(f.x); o.y = f32_to_bf16_rne(f.y);
    o.z = f32_to_bf16_rne(f.z); o.w = f32_to_bf16_rne(f.w);
  } else {
    o = make_ushort4(0, 0, 0, 0);
  }
  reinterpret_cast<ushort4*>(hbf)[g] = o;
}

// -------- kernel 3: WMMA GEMM sweep over a V-chunk + online softmax stats --------
// grid = (rowGroup = 64, chunk = 32), block = 256 (8 waves).
// Block owns 64 rows (4 WMMA row tiles resident in LDS for the full K=2048).
// Each wave owns 1 column tile at a time and 4 row-tile accumulators, so every
// B fragment fetched from L2 feeds 4 WMMAs. Softmax stats are kept per-lane
// (per column subset) and tree-merged once at the end — no bpermute in the
// steady state.
__global__ __launch_bounds__(256)
void flce_gemm_lse_partial(const unsigned short* __restrict__ wbf,
                           const unsigned short* __restrict__ hbf,
                           float* __restrict__ pm, float* __restrict__ ps) {
  // A fragments in contiguous-k order (k = 16*Kset + idx): Afrag[tile][kBlock][lane][0..15],
  // lane = (Kset<<4) | row.  4 x 64KB = 256KB LDS.
  __shared__ __align__(32) unsigned short Afrag[4][H / 32][32][16];
  __shared__ float mergeM[8][MBLK];
  __shared__ float mergeS[8][MBLK];

  const int rowGrp  = blockIdx.x;
  const int chunk   = blockIdx.y;
  const int tid     = threadIdx.x;
  const int wave    = tid >> 5;
  const int lane    = tid & 31;
  const int rowBase = rowGrp * MBLK;

  // stage + permute A (64 x 2048 bf16 = 256KB) into fragment order
  for (int e = tid; e < MBLK * H; e += 256) {
    const int row64 = e >> 11;           // e / H : 0..63
    const int k     = e & (H - 1);
    const unsigned short val = hbf[(size_t)(rowBase + row64) * H + k];
    const int t   = row64 >> 4;
    const int row = row64 & 15;
    const int kb  = k >> 5;
    const int kk  = k & 31;
    const int set = kk >> 4;                       // K in {16..31} -> set1
    const int idx = kk & 15;                       // position inside v16
    Afrag[t][kb][(set << 4) | row][idx] = val;
  }
  __syncthreads();

  // per-lane online stats: [row tile][r], covering this lane's columns only
  float m_run[4][8], s_run[4][8];
#pragma unroll
  for (int t = 0; t < 4; ++t)
#pragma unroll
    for (int r = 0; r < 8; ++r) { m_run[t][r] = -__builtin_inff(); s_run[t][r] = 0.0f; }

  const int colInTile = lane & 15;          // B-matrix column owned by this lane
  const int koff      = (lane >> 4) * 16;   // lanes 16-31 hold K=16..31 half

  for (int ctw = 0; ctw < COLTILES_PER_CHUNK / 8; ++ctw) {
    const int ct = ctw * 8 + wave;
    const size_t col = (size_t)chunk * CHUNKV + (size_t)ct * 16 + colInTile;
    const unsigned short* bptr = wbf + col * H + koff;
    __builtin_prefetch(bptr + 16 * H, 0, 1);  // next column tile

    v8f c0 = {}, c1 = {}, c2 = {}, c3 = {};
#pragma unroll 2
    for (int k = 0; k < H; k += 32) {
      v16bf b  = *reinterpret_cast<const v16bf*>(bptr + k);                    // 2x global_load_b128
      v16bf a0 = *reinterpret_cast<const v16bf*>(&Afrag[0][k >> 5][lane][0]);  // ds_load_b128 x2
      v16bf a1 = *reinterpret_cast<const v16bf*>(&Afrag[1][k >> 5][lane][0]);
      v16bf a2 = *reinterpret_cast<const v16bf*>(&Afrag[2][k >> 5][lane][0]);
      v16bf a3 = *reinterpret_cast<const v16bf*>(&Afrag[3][k >> 5][lane][0]);
      c0 = __builtin_amdgcn_wmma_f32_16x16x32_bf16(false, a0, false, b, (short)0, c0, false, false);
      c1 = __builtin_amdgcn_wmma_f32_16x16x32_bf16(false, a1, false, b, (short)0, c1, false, false);
      c2 = __builtin_amdgcn_wmma_f32_16x16x32_bf16(false, a2, false, b, (short)0, c2, false, false);
      c3 = __builtin_amdgcn_wmma_f32_16x16x32_bf16(false, a3, false, b, (short)0, c3, false, false);
    }

    // fold the 4x(16x16) logit tiles into per-lane online stats (no shuffles)
#pragma unroll
    for (int r = 0; r < 8; ++r) {
      online_merge(c0[r], m_run[0][r], s_run[0][r]);
      online_merge(c1[r], m_run[1][r], s_run[1][r]);
      online_merge(c2[r], m_run[2][r], s_run[2][r]);
      online_merge(c3[r], m_run[3][r], s_run[3][r]);
    }
  }

  // one-time cross-lane tree merge within each 16-lane half (columns of a tile)
#pragma unroll
  for (int t = 0; t < 4; ++t)
#pragma unroll
    for (int r = 0; r < 8; ++r) {
      float m = m_run[t][r], s = s_run[t][r];
      for (int off = 8; off >= 1; off >>= 1) {
        const float om = __shfl_xor(m, off, 16);
        const float os = __shfl_xor(s, off, 16);
        pair_merge(m, s, om, os);
      }
      m_run[t][r] = m; s_run[t][r] = s;
    }

  // C layout: VGPR r -> row r (lanes 0-15) / row r+8 (lanes 16-31)
  if (lane == 0) {
#pragma unroll
    for (int t = 0; t < 4; ++t)
#pragma unroll
      for (int r = 0; r < 8; ++r) {
        mergeM[wave][t * 16 + r] = m_run[t][r];
        mergeS[wave][t * 16 + r] = s_run[t][r];
      }
  } else if (lane == 16) {
#pragma unroll
    for (int t = 0; t < 4; ++t)
#pragma unroll
      for (int r = 0; r < 8; ++r) {
        mergeM[wave][t * 16 + 8 + r] = m_run[t][r];
        mergeS[wave][t * 16 + 8 + r] = s_run[t][r];
      }
  }
  __syncthreads();

  // merge the 8 waves' (disjoint column sets) stats per row, write partials
  if (tid < MBLK) {
    float m = -__builtin_inff(), s = 0.0f;
    for (int w = 0; w < 8; ++w)
      pair_merge(m, s, mergeM[w][tid], mergeS[w][tid]);
    pm[(size_t)(rowBase + tid) * NCHUNK + chunk] = m;
    ps[(size_t)(rowBase + tid) * NCHUNK + chunk] = s;
  }
}

// -------- kernel 4: reduce chunk partials -> per-row logsumexp --------
__global__ __launch_bounds__(256)
void flce_lse_reduce(const float* __restrict__ pm, const float* __restrict__ ps,
                     float* __restrict__ lse) {
  const int row = blockIdx.x * 256 + threadIdx.x;
  if (row >= NPAD) return;
  float m = -__builtin_inff(), s = 0.0f;
  for (int c = 0; c < NCHUNK; ++c)
    pair_merge(m, s, pm[(size_t)row * NCHUNK + c], ps[(size_t)row * NCHUNK + c]);
  lse[row] = m + __logf(s);
}

// -------- kernel 5: target logit (fp32 dot) + loss accumulation --------
__global__ __launch_bounds__(256)
void flce_target_loss(const float* __restrict__ hidden,
                      const float* __restrict__ weight,
                      const int*   __restrict__ labels,
                      const float* __restrict__ lse,
                      float* __restrict__ scal) {
  const int wave = threadIdx.x >> 5, lane = threadIdx.x & 31;
  const int row = blockIdx.x * 8 + wave;
  if (row >= NROWS) return;
  const int lab   = labels[row + 1];                 // shifted labels
  const bool valid = (lab != IGNORE_INDEX);
  const int slab  = lab < 0 ? 0 : (lab > V - 1 ? V - 1 : lab);
  const float* hrow = hidden + (size_t)row  * H;
  const float* wrow = weight + (size_t)slab * H;
  float acc = 0.0f;
  for (int k = lane; k < H; k += 32)
    acc = fmaf(hrow[k], wrow[k], acc);
  for (int off = 16; off >= 1; off >>= 1)
    acc += __shfl_xor(acc, off, 32);
  if (lane == 0 && valid) {
    atomicAdd(&scal[0], lse[row] - acc);
    atomicAdd(&scal[1], 1.0f);
  }
}

// -------- kernel 6: finalize scalar loss --------
__global__ void flce_finalize(const float* __restrict__ scal, float* __restrict__ out) {
  out[0] = scal[0] / fmaxf(scal[1], 1.0f);
}

extern "C" void kernel_launch(void* const* d_in, const int* in_sizes, int n_in,
                              void* d_out, int out_size, void* d_ws, size_t ws_size,
                              hipStream_t stream) {
  (void)in_sizes; (void)n_in; (void)out_size; (void)ws_size;
  const float* hidden = (const float*)d_in[0];
  const float* weight = (const float*)d_in[1];
  const int*   labels = (const int*)d_in[2];
  float* out = (float*)d_out;

  char* ws = (char*)d_ws;
  unsigned short* wbf = (unsigned short*)ws; ws += (size_t)V * H * sizeof(unsigned short);
  unsigned short* hbf = (unsigned short*)ws; ws += (size_t)NPAD * H * sizeof(unsigned short);
  float* pm   = (float*)ws;                  ws += (size_t)NPAD * NCHUNK * sizeof(float);
  float* ps   = (float*)ws;                  ws += (size_t)NPAD * NCHUNK * sizeof(float);
  float* lse  = (float*)ws;                  ws += (size_t)NPAD * sizeof(float);
  float* scal = (float*)ws;

  flce_init<<<1, 32, 0, stream>>>(scal);
  flce_cvt_weight<<<8192, 256, 0, stream>>>(weight, wbf);
  flce_cvt_hidden<<<(NPAD * H / 4) / 256, 256, 0, stream>>>(hidden, hbf);
  flce_gemm_lse_partial<<<dim3(ROWGRPS, NCHUNK), 256, 0, stream>>>(wbf, hbf, pm, ps);
  flce_lse_reduce<<<NPAD / 256, 256, 0, stream>>>(pm, ps, lse);
  flce_target_loss<<<(NROWS + 7) / 8, 256, 0, stream>>>(hidden, weight, labels, lse, scal);
  flce_finalize<<<1, 1, 0, stream>>>(scal, out);
}
